// LearnableSpectralSinkhorn_73031623901726
// MI455X (gfx1250) — compile-verified, matrix operand order checked by reading, not run on hardware
//
#include <hip/hip_runtime.h>

// ---------------------------------------------------------------------------
// LearnableSpectralSinkhorn for MI455X (gfx1250, wave32, WMMA, async-LDS)
//
//   cvt x/up_w/down_w -> bf16
//   sinkhorn (log-space, 10 iters) -> W (bf16)
//   GEMM1: freq(f32) = xb @ upb^T              (bf16 WMMA, f32 accum)
//   mag   : magb(bf16) = sqrt(re^2+im^2+1e-8)
//   GEMM2: proc = magb @ Wb^T ; epilogue: s=proc/(mag+1e-6);
//          freqnb(bf16) = freq * s (both halves)
//   GEMM3: out(f32) = freqnb @ downb^T
// ---------------------------------------------------------------------------

#define DIMD   1024
#define DIMF   2048
#define MROWS  16384          // 4 * 4096
#define TEMP_INV 8.0f         // 1 / 0.125

#define BM 128
#define BN 128
#define BK 32
#define LDT 40                // padded LDS row (bf16 elems): 80B stride

typedef __bf16 bf16_t;
typedef bf16_t v16bf __attribute__((ext_vector_type(16)));
typedef bf16_t v8bf  __attribute__((ext_vector_type(8)));
typedef float  v8f   __attribute__((ext_vector_type(8)));
typedef int    v4i   __attribute__((ext_vector_type(4)));
typedef __attribute__((address_space(3))) v4i lds_v4i;

#if defined(__has_builtin)
#if __has_builtin(__builtin_amdgcn_global_load_async_to_lds_b128)
#define USE_ASYNC_LDS 1
#endif
#endif

union FragAB { v16bf v; v8bf h[2]; };

// ------------------------------- small kernels -----------------------------

__global__ void k_cvt_f32_bf16(const float* __restrict__ in, bf16_t* __restrict__ out, int n) {
    int i = blockIdx.x * blockDim.x + threadIdx.x;
    if (i < n) out[i] = (bf16_t)in[i];
}

__global__ void k_sink_init(const float* __restrict__ w, float* __restrict__ logP, int n) {
    int i = blockIdx.x * blockDim.x + threadIdx.x;
    if (i < n) logP[i] = w[i] * TEMP_INV;
}

// blocks 0..1023 -> row logsumexp, blocks 1024..2047 -> col logsumexp
__global__ __launch_bounds__(256) void k_sink_lse(const float* __restrict__ logP,
                                                  float* __restrict__ rbuf,
                                                  float* __restrict__ cbuf) {
    __shared__ float red[256];
    const int tid = threadIdx.x;
    const int bid = blockIdx.x;
    const bool isRow = (bid < DIMD);
    const int line = isRow ? bid : (bid - DIMD);

    float mx = -3.4e38f;
    for (int j = tid; j < DIMD; j += 256) {
        float v = isRow ? logP[line * DIMD + j] : logP[j * DIMD + line];
        mx = fmaxf(mx, v);
    }
    red[tid] = mx; __syncthreads();
    for (int s = 128; s > 0; s >>= 1) {
        if (tid < s) red[tid] = fmaxf(red[tid], red[tid + s]);
        __syncthreads();
    }
    mx = red[0]; __syncthreads();

    float sum = 0.0f;
    for (int j = tid; j < DIMD; j += 256) {
        float v = isRow ? logP[line * DIMD + j] : logP[j * DIMD + line];
        sum += __expf(v - mx);
    }
    red[tid] = sum; __syncthreads();
    for (int s = 128; s > 0; s >>= 1) {
        if (tid < s) red[tid] += red[tid + s];
        __syncthreads();
    }
    if (tid == 0) {
        float lse = mx + __logf(red[0]);
        if (isRow) rbuf[line] = lse; else cbuf[line] = lse;
    }
}

__global__ void k_sink_update(float* __restrict__ logP,
                              const float* __restrict__ rbuf,
                              const float* __restrict__ cbuf) {
    int idx = blockIdx.x * blockDim.x + threadIdx.x;
    if (idx < DIMD * DIMD) {
        int i = idx >> 10, j = idx & (DIMD - 1);
        logP[idx] -= 0.5f * (rbuf[i] + cbuf[j]);
    }
}

__global__ void k_sink_exp(const float* __restrict__ logP, bf16_t* __restrict__ Wb, int n) {
    int i = blockIdx.x * blockDim.x + threadIdx.x;
    if (i < n) Wb[i] = (bf16_t)__expf(logP[i]);
}

__global__ void k_mag(const float* __restrict__ freq, bf16_t* __restrict__ magb, int n) {
    int idx = blockIdx.x * blockDim.x + threadIdx.x;
    if (idx < n) {
        int r = idx >> 10, d = idx & (DIMD - 1);
        float re = freq[(size_t)r * DIMF + d];
        float im = freq[(size_t)r * DIMF + DIMD + d];
        magb[idx] = (bf16_t)sqrtf(re * re + im * im + 1e-8f);
    }
}

// ------------------------------- WMMA GEMM ---------------------------------
// C[M,N] = A[M,K](bf16) * B[N,K](bf16)^T, f32 accumulate via WMMA.
// EPI 0: store f32 into C
// EPI 1: s = acc/(mag+1e-6); freqnb[row, col(+1024)] = bf16(freq * s)

__device__ __forceinline__ void stage_async(const bf16_t* __restrict__ A,
                                            const bf16_t* __restrict__ B,
                                            bf16_t* sA, bf16_t* sB,
                                            int mBlk, int nBlk, int ldA, int ldB,
                                            int k0, int tid) {
#ifdef USE_ASYNC_LDS
#pragma unroll
    for (int i = 0; i < 2; ++i) {                  // A tile: 128x32 bf16 = 512 x 16B
        int idx = tid + i * 256;
        int r = idx >> 2;
        int c = (idx & 3) << 3;
        const bf16_t* g = A + (size_t)(mBlk + r) * ldA + k0 + c;
        bf16_t* l = sA + r * LDT + c;
        __builtin_amdgcn_global_load_async_to_lds_b128(
            (v4i*)g,
            (lds_v4i*)(unsigned int)(unsigned long long)(void*)l,
            0, 0);
    }
#pragma unroll
    for (int i = 0; i < 2; ++i) {                  // B tile: 128x32 bf16 = 512 x 16B
        int idx = tid + i * 256;
        int r = idx >> 2;
        int c = (idx & 3) << 3;
        const bf16_t* g = B + (size_t)(nBlk + r) * ldB + k0 + c;
        bf16_t* l = sB + r * LDT + c;
        __builtin_amdgcn_global_load_async_to_lds_b128(
            (v4i*)g,
            (lds_v4i*)(unsigned int)(unsigned long long)(void*)l,
            0, 0);
    }
#else
    (void)A; (void)B; (void)sA; (void)sB; (void)mBlk; (void)nBlk;
    (void)ldA; (void)ldB; (void)k0; (void)tid;
#endif
}

template <int EPI>
__global__ __launch_bounds__(256) void gemm_bf16_wmma(
    const bf16_t* __restrict__ A, const bf16_t* __restrict__ B,
    float* __restrict__ C, int K, int ldA, int ldC,
    const bf16_t* __restrict__ magb, const float* __restrict__ freqf,
    bf16_t* __restrict__ freqnb)
{
    __shared__ bf16_t sA[2][BM * LDT];
    __shared__ bf16_t sB[2][BN * LDT];

    const int tid   = threadIdx.x;
    const int lane  = tid & 31;
    const int wave  = tid >> 5;          // 8 waves
    const int waveM = wave & 1;          // 2 along M -> 64 rows per wave
    const int waveN = wave >> 1;         // 4 along N -> 32 cols per wave
    const int mBlk  = blockIdx.y * BM;
    const int nBlk  = blockIdx.x * BN;

    v8f acc[4][2];
    const v8f zero8 = {0.f, 0.f, 0.f, 0.f, 0.f, 0.f, 0.f, 0.f};
#pragma unroll
    for (int mi = 0; mi < 4; ++mi)
#pragma unroll
        for (int ni = 0; ni < 2; ++ni) acc[mi][ni] = zero8;

    const int halfK = (lane >> 4) << 3;  // 0 (lanes 0-15) or 8 (lanes 16-31)
    const int rsel  = lane & 15;
    const int kTiles = K / BK;
    int cur = 0;

#ifdef USE_ASYNC_LDS
    stage_async(A, B, sA[0], sB[0], mBlk, nBlk, ldA, K, 0, tid);
#else
    v8bf ra[2], rb[2];
    {
#pragma unroll
        for (int i = 0; i < 2; ++i) {
            int idx = tid + i * 256;
            int r = idx >> 2, c = (idx & 3) << 3;
            ra[i] = *(const v8bf*)(A + (size_t)(mBlk + r) * ldA + c);
            rb[i] = *(const v8bf*)(B + (size_t)(nBlk + r) * K + c);
        }
    }
#endif

    for (int kt = 0; kt < kTiles; ++kt) {
        const bool more = (kt + 1 < kTiles);
#ifdef USE_ASYNC_LDS
        if (more) {
            stage_async(A, B, sA[cur ^ 1], sB[cur ^ 1], mBlk, nBlk, ldA, K,
                        (kt + 1) * BK, tid);
            asm volatile("s_wait_asynccnt 0x4" ::: "memory");
        } else {
            asm volatile("s_wait_asynccnt 0x0" ::: "memory");
        }
        __syncthreads();
#else
        // store staged regs into current buffer
#pragma unroll
        for (int i = 0; i < 2; ++i) {
            int idx = tid + i * 256;
            int r = idx >> 2, c = (idx & 3) << 3;
            *(v8bf*)&sA[cur][r * LDT + c] = ra[i];
            *(v8bf*)&sB[cur][r * LDT + c] = rb[i];
        }
        __syncthreads();
        if (more) {
            const int k1 = (kt + 1) * BK;
#pragma unroll
            for (int i = 0; i < 2; ++i) {
                int idx = tid + i * 256;
                int r = idx >> 2, c = (idx & 3) << 3;
                ra[i] = *(const v8bf*)(A + (size_t)(mBlk + r) * ldA + k1 + c);
                rb[i] = *(const v8bf*)(B + (size_t)(nBlk + r) * K + k1 + c);
            }
        }
#endif

        // fragments per ISA 16-bit A 16x32 layout:
        // lane<16 : K [0..8) and [16..24) ; lane>=16 : K [8..16) and [24..32)
        FragAB aF[4];
#pragma unroll
        for (int mi = 0; mi < 4; ++mi) {
            const bf16_t* base = &sA[cur][(waveM * 64 + mi * 16 + rsel) * LDT];
            aF[mi].h[0] = *(const v8bf*)(base + halfK);
            aF[mi].h[1] = *(const v8bf*)(base + 16 + halfK);
        }
        FragAB bF[2];
#pragma unroll
        for (int ni = 0; ni < 2; ++ni) {
            const bf16_t* base = &sB[cur][(waveN * 32 + ni * 16 + rsel) * LDT];
            bF[ni].h[0] = *(const v8bf*)(base + halfK);
            bF[ni].h[1] = *(const v8bf*)(base + 16 + halfK);
        }

#pragma unroll
        for (int mi = 0; mi < 4; ++mi)
#pragma unroll
            for (int ni = 0; ni < 2; ++ni)
                acc[mi][ni] = __builtin_amdgcn_wmma_f32_16x16x32_bf16(
                    false, aF[mi].v, false, bF[ni].v,
                    (short)0, acc[mi][ni], false, false);
        __syncthreads();
        cur ^= 1;
    }

    // epilogue: C/D layout -> col = lane&15, row = 8*(lane>>4) + vgpr
    const int rhi = (lane >> 4) << 3;
#pragma unroll
    for (int mi = 0; mi < 4; ++mi) {
#pragma unroll
        for (int ni = 0; ni < 2; ++ni) {
#pragma unroll
            for (int r = 0; r < 8; ++r) {
                const int row = mBlk + waveM * 64 + mi * 16 + rhi + r;
                const int col = nBlk + waveN * 32 + ni * 16 + rsel;
                const float v = acc[mi][ni][r];
                if (EPI == 0) {
                    C[(size_t)row * ldC + col] = v;
                } else {
                    const float m = (float)magb[(size_t)row * DIMD + col];
                    const float s = v / (m + 1e-6f);
                    const float re = freqf[(size_t)row * DIMF + col];
                    const float im = freqf[(size_t)row * DIMF + DIMD + col];
                    freqnb[(size_t)row * DIMF + col]        = (bf16_t)(re * s);
                    freqnb[(size_t)row * DIMF + DIMD + col] = (bf16_t)(im * s);
                }
            }
        }
    }
}

// ------------------------------- launcher ----------------------------------

extern "C" void kernel_launch(void* const* d_in, const int* in_sizes, int n_in,
                              void* d_out, int out_size, void* d_ws, size_t ws_size,
                              hipStream_t stream) {
    (void)in_sizes; (void)n_in; (void)out_size; (void)ws_size;

    const float* x      = (const float*)d_in[0];   // [4,4096,1024]
    const float* up_w   = (const float*)d_in[1];   // [2048,1024]
    const float* down_w = (const float*)d_in[2];   // [1024,2048]
    const float* sink_w = (const float*)d_in[3];   // [1024,1024]
    float*       out    = (float*)d_out;           // [4,4096,1024]

    char* ws = (char*)d_ws;
    size_t off = 0;
    float*  freq   = (float*)(ws + off);  off += (size_t)MROWS * DIMF * 4;  // 128 MB
    bf16_t* xb     = (bf16_t*)(ws + off); off += (size_t)MROWS * DIMD * 2;  //  32 MB
    bf16_t* magb   = (bf16_t*)(ws + off); off += (size_t)MROWS * DIMD * 2;  //  32 MB
    bf16_t* freqnb = (bf16_t*)(ws + off); off += (size_t)MROWS * DIMF * 2;  //  64 MB
    bf16_t* upb    = (bf16_t*)(ws + off); off += (size_t)DIMF * DIMD * 2;   //   4 MB
    bf16_t* downb  = (bf16_t*)(ws + off); off += (size_t)DIMD * DIMF * 2;   //   4 MB
    bf16_t* Wb     = (bf16_t*)(ws + off); off += (size_t)DIMD * DIMD * 2;   //   2 MB
    float*  logP   = (float*)(ws + off);  off += (size_t)DIMD * DIMD * 4;   //   4 MB
    float*  rbuf   = (float*)(ws + off);  off += 4096;
    float*  cbuf   = (float*)(ws + off);  off += 4096;

    const int nW = DIMF * DIMD;          // 2M elems (up/down weights)
    const int nP = DIMD * DIMD;          // 1M elems (sinkhorn matrix)
    const int nX = MROWS * DIMD;         // 16M elems (x / mag)

    // bf16 operand preparation (bandwidth-bound, L2-resident afterwards)
    k_cvt_f32_bf16<<<(nW + 255) / 256, 256, 0, stream>>>(up_w, upb, nW);
    k_cvt_f32_bf16<<<(nW + 255) / 256, 256, 0, stream>>>(down_w, downb, nW);
    k_cvt_f32_bf16<<<(nX + 255) / 256, 256, 0, stream>>>(x, xb, nX);

    // sinkhorn (log-space, 10 iterations)
    k_sink_init<<<(nP + 255) / 256, 256, 0, stream>>>(sink_w, logP, nP);
    for (int it = 0; it < 10; ++it) {
        k_sink_lse<<<2 * DIMD, 256, 0, stream>>>(logP, rbuf, cbuf);
        k_sink_update<<<(nP + 255) / 256, 256, 0, stream>>>(logP, rbuf, cbuf);
    }
    k_sink_exp<<<(nP + 255) / 256, 256, 0, stream>>>(logP, Wb, nP);

    // GEMM1: freq = x @ up_w^T   (M=16384, N=2048, K=1024)
    gemm_bf16_wmma<0><<<dim3(DIMF / BN, MROWS / BM), 256, 0, stream>>>(
        xb, upb, freq, DIMD, DIMD, DIMF, nullptr, nullptr, nullptr);

    // magnitude -> bf16 operand
    k_mag<<<(nX + 255) / 256, 256, 0, stream>>>(freq, magb, nX);

    // GEMM2: proc = mag @ W^T ; epilogue writes bf16 freq_new (M=16384, N=1024, K=1024)
    gemm_bf16_wmma<1><<<dim3(DIMD / BN, MROWS / BM), 256, 0, stream>>>(
        magb, Wb, nullptr, DIMD, DIMD, 0, magb, freq, freqnb);

    // GEMM3: out = freq_new @ down_w^T (M=16384, N=1024, K=2048)
    gemm_bf16_wmma<0><<<dim3(DIMD / BN, MROWS / BM), 256, 0, stream>>>(
        freqnb, downb, out, DIMF, DIMF, DIMD, nullptr, nullptr, nullptr);
}
